// LSTM_double_54202487276119
// MI455X (gfx1250) — matrix operand order repaired and checked by reading, not compiled, not run on hardware
//
#include <hip/hip_runtime.h>

// ---------------- problem dims ----------------
#define T_STEPS 128
#define BATCH   64
#define NHID    1024
#define EMBD    512
#define NCLS    32000
#define K1      (NHID + EMBD)   // 1536
#define K2      (2 * NHID)      // 2048
#define LSTM_WGS     64         // each owns 16 hidden units
#define LSTM_THREADS 128        // 4 waves = 4 gates

// ---------------- workspace layout (bytes) ----------------
#define OFF_BAR   0
#define OFF_H1    256
#define OFF_H2    (OFF_H1 + 2 * BATCH * NHID * 2)                 // ping-pong bf16
#define OFF_XEMB  (OFF_H2 + 2 * BATCH * NHID * 2)
#define OFF_W1    (OFF_XEMB + T_STEPS * BATCH * EMBD * 2)
#define OFF_W2    (OFF_W1 + 4 * NHID * K1 * 2)
// total ~38.3 MB

// ---------------- d_out offsets (floats) ----------------
#define OUT_H1 (BATCH * NCLS)
#define OUT_H2 (OUT_H1 + BATCH * NHID)
#define OUT_C1 (OUT_H2 + BATCH * NHID)
#define OUT_C2 (OUT_C1 + BATCH * NHID)

typedef __attribute__((ext_vector_type(16))) __bf16 bf16x16;
typedef __attribute__((ext_vector_type(8)))  float  floatx8;
typedef __attribute__((ext_vector_type(4)))  unsigned int uintx4;
typedef __attribute__((ext_vector_type(4)))  float  floatx4;

union BFrag { bf16x16 v; uintx4 u[2]; };

__device__ __forceinline__ float sigm(float x)   { return 1.0f / (1.0f + __expf(-x)); }
__device__ __forceinline__ float tanh_f(float x) { return 1.0f - 2.0f / (__expf(2.0f * x) + 1.0f); }

__device__ __forceinline__ floatx8 wmma_bf16(bf16x16 a, bf16x16 b, floatx8 c) {
  // v_wmma_f32_16x16x32_bf16  (neg_a, A, neg_b, B, c_mod, C, reuse_a, reuse_b)
  return __builtin_amdgcn_wmma_f32_16x16x32_bf16(false, a, false, b, (short)0, c, false, false);
}

// device-wide sense-reversing barrier (persistent kernel, grid assumed resident)
__device__ __forceinline__ void gridBarrier(unsigned* bar, unsigned* lds_sense, unsigned nwg) {
  __syncthreads();
  if (threadIdx.x == 0) {
    unsigned ls = *lds_sense ^ 1u;
    *lds_sense = ls;
    __threadfence();                                   // flush h writes device-scope
    unsigned prev = __hip_atomic_fetch_add(bar, 1u, __ATOMIC_ACQ_REL, __HIP_MEMORY_SCOPE_AGENT);
    if (prev == nwg - 1u) {
      bar[0] = 0u;                                     // reset count before release
      __hip_atomic_store(bar + 1, ls, __ATOMIC_RELEASE, __HIP_MEMORY_SCOPE_AGENT);
    } else {
      while (__hip_atomic_load(bar + 1, __ATOMIC_ACQUIRE, __HIP_MEMORY_SCOPE_AGENT) != ls)
        __builtin_amdgcn_s_sleep(1);
    }
    __threadfence();
  }
  __syncthreads();
}

// ---------------- small helper kernels ----------------
__global__ void init_ws(unsigned* bar, __bf16* hzero, int nelem) {
  if (blockIdx.x == 0 && threadIdx.x == 0) { bar[0] = 0u; bar[1] = 0u; }
  int stride = (int)(gridDim.x * blockDim.x);
  for (int i = blockIdx.x * blockDim.x + threadIdx.x; i < nelem; i += stride)
    hzero[i] = (__bf16)0.0f;
}

__global__ void convert_f32_bf16(const float* __restrict__ src, __bf16* __restrict__ dst, int n) {
  int stride = (int)(gridDim.x * blockDim.x);
  for (int i = blockIdx.x * blockDim.x + threadIdx.x; i < n; i += stride)
    dst[i] = (__bf16)src[i];
}

// Xemb[t][b][e] = bf16( C_emb[ X[b][t] ][e] )
__global__ void embed_gather(const int* __restrict__ X, const float* __restrict__ C,
                             __bf16* __restrict__ xemb) {
  const long total = (long)T_STEPS * BATCH * EMBD;
  const long stride = (long)gridDim.x * blockDim.x;
  for (long i = blockIdx.x * (long)blockDim.x + threadIdx.x; i < total; i += stride) {
    int e = (int)(i % EMBD);
    int b = (int)((i / EMBD) % BATCH);
    int t = (int)(i / ((long)EMBD * BATCH));
    int tok = X[b * T_STEPS + t];
    xemb[i] = (__bf16)C[(size_t)tok * EMBD + e];
  }
}

// ---------------- persistent 2-layer LSTM scan ----------------
// 64 WGs x 128 threads. WG g owns hidden units [16g,16g+16); wave q computes gate q.
// GEMM: A = cat (bf16, row-major), B = torch-layout [out][in] bf16 == native WMMA B frag.
__global__ __launch_bounds__(LSTM_THREADS, 1) void lstm_scan(
    const __bf16* __restrict__ w1f, const __bf16* __restrict__ w1i,
    const __bf16* __restrict__ w1c, const __bf16* __restrict__ w1o,
    const __bf16* __restrict__ w2f, const __bf16* __restrict__ w2i,
    const __bf16* __restrict__ w2c, const __bf16* __restrict__ w2o,
    const float* __restrict__ b1f, const float* __restrict__ b1i,
    const float* __restrict__ b1c, const float* __restrict__ b1o,
    const float* __restrict__ b2f, const float* __restrict__ b2i,
    const float* __restrict__ b2c, const float* __restrict__ b2o,
    const __bf16* __restrict__ xemb,
    __bf16* __restrict__ h1buf, __bf16* __restrict__ h2buf,
    float* __restrict__ dout, unsigned* __restrict__ bar)
{
  __shared__ float s_g[4][BATCH * 16];   // gate pre-activations for this WG's 16 units
  __shared__ float s_c1[BATCH * 16];     // persistent cell state, layer 1
  __shared__ float s_c2[BATCH * 16];     // persistent cell state, layer 2
  __shared__ unsigned s_sense;

  const int wg   = blockIdx.x;
  const int tid  = threadIdx.x;
  const int wave = tid >> 5;             // gate id 0..3 (f,i,c,o)
  const int lane = tid & 31;
  const int nloc   = lane & 15;          // N col / A row within tile
  const int khalfA = (lane >> 4) * 16;   // A fragment K sub-offset
  const int koffB  = (lane >> 4) * 16;   // B fragment K offset
  const int mofs   = (lane >> 4) * 8;    // C/D fragment M offset
  const int nbase  = wg * 16;

  if (tid == 0) s_sense = 0u;
  for (int i = tid; i < BATCH * 16; i += LSTM_THREADS) { s_c1[i] = 0.0f; s_c2[i] = 0.0f; }
  __syncthreads();

  const __bf16* W1 = (wave == 0) ? w1f : (wave == 1) ? w1i : (wave == 2) ? w1c : w1o;
  const __bf16* W2 = (wave == 0) ? w2f : (wave == 1) ? w2i : (wave == 2) ? w2c : w2o;

  for (int t = 0; t < T_STEPS; ++t) {
    const int rb = t & 1, wb = rb ^ 1;
    const __bf16* h1r = h1buf + rb * (BATCH * NHID);
    __bf16*       h1w = h1buf + wb * (BATCH * NHID);
    const __bf16* h2r = h2buf + rb * (BATCH * NHID);
    __bf16*       h2w = h2buf + wb * (BATCH * NHID);
    const __bf16* xt  = xemb + (size_t)t * (BATCH * EMBD);
    const bool last = (t == T_STEPS - 1);

    // ---------- layer 1 GEMM: [64,K1] x [K1,16] for this wave's gate ----------
    {
      floatx8 acc[4];
      acc[0] = (floatx8)0.0f; acc[1] = (floatx8)0.0f;
      acc[2] = (floatx8)0.0f; acc[3] = (floatx8)0.0f;
      const __bf16* wbase = W1 + (size_t)(nbase + nloc) * K1;
      for (int kk = 0; kk < K1; kk += 32) {
        BFrag bf;
        const __bf16* wrow = wbase + kk + koffB;
        bf.u[0] = *(const uintx4*)(wrow);
        bf.u[1] = *(const uintx4*)(wrow + 8);
        const __bf16* asrc; int ld, col;
        if (kk < NHID) { asrc = h1r; ld = NHID; col = kk; }
        else           { asrc = xt;  ld = EMBD; col = kk - NHID; }
        #pragma unroll
        for (int mt = 0; mt < 4; ++mt) {
          BFrag af;
          const __bf16* arow = asrc + (size_t)(mt * 16 + nloc) * ld + col + khalfA;
          af.u[0] = *(const uintx4*)(arow);
          af.u[1] = *(const uintx4*)(arow + 8);
          acc[mt] = wmma_bf16(af.v, bf.v, acc[mt]);
        }
      }
      #pragma unroll
      for (int mt = 0; mt < 4; ++mt)
        #pragma unroll
        for (int r = 0; r < 8; ++r)
          s_g[wave][(mt * 16 + mofs + r) * 16 + nloc] = acc[mt][r];
    }
    __syncthreads();

    // ---------- layer 1 cell update ----------
    for (int idx = tid; idx < BATCH * 16; idx += LSTM_THREADS) {
      int m = idx >> 4, n = idx & 15, col = nbase + n;
      float f  = sigm  (s_g[0][idx] + b1f[col]);
      float i  = sigm  (s_g[1][idx] + b1i[col]);
      float cc = tanh_f(s_g[2][idx] + b1c[col]);
      float o  = sigm  (s_g[3][idx] + b1o[col]);
      float c  = f * s_c1[idx] + i * cc;
      s_c1[idx] = c;
      float h  = o * tanh_f(c);
      h1w[(size_t)m * NHID + col] = (__bf16)h;
      if (last) {
        dout[OUT_H1 + (size_t)m * NHID + col] = h;
        dout[OUT_C1 + (size_t)m * NHID + col] = c;
      }
    }
    gridBarrier(bar, &s_sense, (unsigned)gridDim.x);

    // ---------- layer 2 GEMM: [64,K2] x [K2,16] ----------
    {
      floatx8 acc[4];
      acc[0] = (floatx8)0.0f; acc[1] = (floatx8)0.0f;
      acc[2] = (floatx8)0.0f; acc[3] = (floatx8)0.0f;
      const __bf16* wbase = W2 + (size_t)(nbase + nloc) * K2;
      for (int kk = 0; kk < K2; kk += 32) {
        BFrag bf;
        const __bf16* wrow = wbase + kk + koffB;
        bf.u[0] = *(const uintx4*)(wrow);
        bf.u[1] = *(const uintx4*)(wrow + 8);
        const __bf16* asrc; int col;
        if (kk < NHID) { asrc = h2r; col = kk; }
        else           { asrc = h1w; col = kk - NHID; }   // h1 written this step
        #pragma unroll
        for (int mt = 0; mt < 4; ++mt) {
          BFrag af;
          const __bf16* arow = asrc + (size_t)(mt * 16 + nloc) * NHID + col + khalfA;
          af.u[0] = *(const uintx4*)(arow);
          af.u[1] = *(const uintx4*)(arow + 8);
          acc[mt] = wmma_bf16(af.v, bf.v, acc[mt]);
        }
      }
      #pragma unroll
      for (int mt = 0; mt < 4; ++mt)
        #pragma unroll
        for (int r = 0; r < 8; ++r)
          s_g[wave][(mt * 16 + mofs + r) * 16 + nloc] = acc[mt][r];
    }
    __syncthreads();

    // ---------- layer 2 cell update ----------
    for (int idx = tid; idx < BATCH * 16; idx += LSTM_THREADS) {
      int m = idx >> 4, n = idx & 15, col = nbase + n;
      float f  = sigm  (s_g[0][idx] + b2f[col]);
      float i  = sigm  (s_g[1][idx] + b2i[col]);
      float cc = tanh_f(s_g[2][idx] + b2c[col]);
      float o  = sigm  (s_g[3][idx] + b2o[col]);
      float c  = f * s_c2[idx] + i * cc;
      s_c2[idx] = c;
      float h  = o * tanh_f(c);
      h2w[(size_t)m * NHID + col] = (__bf16)h;
      if (last) {
        dout[OUT_H2 + (size_t)m * NHID + col] = h;
        dout[OUT_C2 + (size_t)m * NHID + col] = c;
      }
    }
    gridBarrier(bar, &s_sense, (unsigned)gridDim.x);
  }
}

// ---------------- output projection: [64,1024] bf16 @ W_w.T (fp32->bf16 in-reg) ----------------
__global__ __launch_bounds__(256, 1) void proj_kernel(
    const __bf16* __restrict__ h2, const float* __restrict__ Ww,
    const float* __restrict__ Wb, float* __restrict__ out)
{
  const int tid  = threadIdx.x;
  const int lane = tid & 31;
  const int wid    = blockIdx.x * (blockDim.x >> 5) + (tid >> 5);
  const int nwaves = gridDim.x * (blockDim.x >> 5);
  const int nloc   = lane & 15;
  const int khalfA = (lane >> 4) * 16;
  const int koffB  = (lane >> 4) * 16;
  const int mofs   = (lane >> 4) * 8;

  for (int strip = wid; strip < NCLS / 16; strip += nwaves) {
    const int nbase = strip * 16;
    floatx8 acc[4];
    acc[0] = (floatx8)0.0f; acc[1] = (floatx8)0.0f;
    acc[2] = (floatx8)0.0f; acc[3] = (floatx8)0.0f;
    const float* wbase = Ww + (size_t)(nbase + nloc) * NHID;
    for (int kk = 0; kk < NHID; kk += 32) {
      const float* wrow = wbase + kk + koffB;
      floatx4 f0 = *(const floatx4*)(wrow + 0);
      floatx4 f1 = *(const floatx4*)(wrow + 4);
      floatx4 f2 = *(const floatx4*)(wrow + 8);
      floatx4 f3 = *(const floatx4*)(wrow + 12);
      BFrag bf;
      #pragma unroll
      for (int e = 0; e < 4; ++e) {
        bf.v[e]      = (__bf16)f0[e];
        bf.v[4 + e]  = (__bf16)f1[e];
        bf.v[8 + e]  = (__bf16)f2[e];
        bf.v[12 + e] = (__bf16)f3[e];
      }
      #pragma unroll
      for (int mt = 0; mt < 4; ++mt) {
        BFrag af;
        const __bf16* arow = h2 + (size_t)(mt * 16 + nloc) * NHID + kk + khalfA;
        af.u[0] = *(const uintx4*)(arow);
        af.u[1] = *(const uintx4*)(arow + 8);
        acc[mt] = wmma_bf16(af.v, bf.v, acc[mt]);
      }
    }
    const float bias = Wb[nbase + nloc];
    #pragma unroll
    for (int mt = 0; mt < 4; ++mt)
      #pragma unroll
      for (int r = 0; r < 8; ++r) {
        int m = mt * 16 + mofs + r;
        out[(size_t)m * NCLS + nbase + nloc] = acc[mt][r] + bias;
      }
  }
}

// ---------------- host launcher ----------------
extern "C" void kernel_launch(void* const* d_in, const int* in_sizes, int n_in,
                              void* d_out, int out_size, void* d_ws, size_t ws_size,
                              hipStream_t stream) {
  (void)in_sizes; (void)n_in; (void)out_size; (void)ws_size;
  char* ws = (char*)d_ws;
  unsigned* bar   = (unsigned*)(ws + OFF_BAR);
  __bf16* h1buf   = (__bf16*)(ws + OFF_H1);
  __bf16* h2buf   = (__bf16*)(ws + OFF_H2);
  __bf16* xemb    = (__bf16*)(ws + OFF_XEMB);
  __bf16* w1      = (__bf16*)(ws + OFF_W1);   // 4 x [1024 x 1536] (f,i,c,o)
  __bf16* w2      = (__bf16*)(ws + OFF_W2);   // 4 x [1024 x 2048]

  const int W1SZ = NHID * K1;  // per gate
  const int W2SZ = NHID * K2;

  // zero barrier vars + both h ping-pong buffers (h1 and h2 are contiguous)
  init_ws<<<64, 256, 0, stream>>>(bar, h1buf, 2 * 2 * BATCH * NHID);

  // convert layer weights fp32 -> bf16 (torch [out][in] layout == WMMA B layout)
  convert_f32_bf16<<<512, 256, 0, stream>>>((const float*)d_in[2],  w1 + 0 * (size_t)W1SZ, W1SZ);
  convert_f32_bf16<<<512, 256, 0, stream>>>((const float*)d_in[4],  w1 + 1 * (size_t)W1SZ, W1SZ);
  convert_f32_bf16<<<512, 256, 0, stream>>>((const float*)d_in[6],  w1 + 2 * (size_t)W1SZ, W1SZ);
  convert_f32_bf16<<<512, 256, 0, stream>>>((const float*)d_in[8],  w1 + 3 * (size_t)W1SZ, W1SZ);
  convert_f32_bf16<<<512, 256, 0, stream>>>((const float*)d_in[10], w2 + 0 * (size_t)W2SZ, W2SZ);
  convert_f32_bf16<<<512, 256, 0, stream>>>((const float*)d_in[12], w2 + 1 * (size_t)W2SZ, W2SZ);
  convert_f32_bf16<<<512, 256, 0, stream>>>((const float*)d_in[14], w2 + 2 * (size_t)W2SZ, W2SZ);
  convert_f32_bf16<<<512, 256, 0, stream>>>((const float*)d_in[16], w2 + 3 * (size_t)W2SZ, W2SZ);

  // embedding gather + bf16 convert, time-major
  embed_gather<<<1024, 256, 0, stream>>>((const int*)d_in[0], (const float*)d_in[1], xemb);

  // persistent scan
  lstm_scan<<<LSTM_WGS, LSTM_THREADS, 0, stream>>>(
      w1 + 0 * (size_t)W1SZ, w1 + 1 * (size_t)W1SZ, w1 + 2 * (size_t)W1SZ, w1 + 3 * (size_t)W1SZ,
      w2 + 0 * (size_t)W2SZ, w2 + 1 * (size_t)W2SZ, w2 + 2 * (size_t)W2SZ, w2 + 3 * (size_t)W2SZ,
      (const float*)d_in[3],  (const float*)d_in[5],  (const float*)d_in[7],  (const float*)d_in[9],
      (const float*)d_in[11], (const float*)d_in[13], (const float*)d_in[15], (const float*)d_in[17],
      xemb, h1buf, h2buf, (float*)d_out, bar);

  // final h2 lives in ping-pong slot ((T-1)&1)^1
  const int finalIdx = ((T_STEPS - 1) & 1) ^ 1;
  const __bf16* h2fin = h2buf + (size_t)finalIdx * (BATCH * NHID);

  // projection: 2000 strips of 16 cols; 250 WGs x 8 waves = 1 strip/wave
  proj_kernel<<<250, 256, 0, stream>>>(h2fin, (const float*)d_in[18],
                                       (const float*)d_in[19], (float*)d_out);
}
